// SoftPatternClassifier_24146306138518
// MI455X (gfx1250) — matrix operand order repaired and checked by reading, not compiled, not run on hardware
//
#include <hip/hip_runtime.h>

typedef __attribute__((ext_vector_type(16))) _Float16 v16h;
typedef __attribute__((ext_vector_type(8)))  _Float16 v8h;
typedef __attribute__((ext_vector_type(8)))  float    v8f;

#define PDIM 300
#define LDIM 7
#define BDIM 64
#define TDIM 256
#define VDIM 10000
#define EDIM 300
#define HDIM 256
#define CDIM 5
#define DDIM 4200            // P * NUM_DIAGS * L
#define KPAD 320             // E padded to multiple of 32
#define MPAD 4224            // D padded to multiple of 32 (2 M-tiles per wave)
#define NPAD 10048           // V padded to multiple of 64 (4 N-tiles per wave)
#define NSTRIPS 157          // NPAD / 64
#define MSTRIPS 132          // MPAD / 32
#define NEGV  (-1000000000.0f)

// ---------------------------------------------------------------------------
// Pack diags (D x E, f32) -> Ah (MPAD x KPAD, f16, zero padded)
// ---------------------------------------------------------------------------
__global__ void __launch_bounds__(256) pack_a(const float* __restrict__ diags,
                                              _Float16* __restrict__ Ah) {
    int idx = blockIdx.x * blockDim.x + threadIdx.x;
    if (idx >= MPAD * KPAD) return;
    int m = idx / KPAD, k = idx % KPAD;
    float x = (m < DDIM && k < EDIM) ? diags[m * EDIM + k] : 0.0f;
    Ah[idx] = (_Float16)x;
}

// ---------------------------------------------------------------------------
// Pack local_embeddings (E x V, f32) -> BhT (NPAD x KPAD, f16, zero padded).
// Transposed so each WMMA B-fragment lane reads two contiguous runs of 8 K values.
// ---------------------------------------------------------------------------
__global__ void __launch_bounds__(256) pack_b(const float* __restrict__ emb,
                                              _Float16* __restrict__ BhT) {
    int idx = blockIdx.x * blockDim.x + threadIdx.x;
    if (idx >= NPAD * KPAD) return;
    int v = idx / KPAD, k = idx % KPAD;
    float x = (k < EDIM && v < VDIM) ? emb[k * VDIM + v] : 0.0f;
    BhT[idx] = (_Float16)x;
}

// ---------------------------------------------------------------------------
// Fragment loaders (layouts per CDNA5 ISA 7.12.2, wave32):
//   A 16x32 f16 : lane<16 -> M=lane, elems 0..7 = K 0..7,  8..15 = K 16..23
//                 lane>=16-> M=lane-16, elems = K 8..15 / 24..31
//   B 32x16 f16 : lane%16 = N col, same K order as A (K 0..7/16..23 vs 8..15/24..31)
// Pointers are pre-offset per lane so each call is two contiguous 16B loads.
// ---------------------------------------------------------------------------
__device__ inline v16h load_frag_a(const _Float16* p) {
    v8h lo = *(const v8h*)(p);
    v8h hi = *(const v8h*)(p + 16);
    v16h r;
#pragma unroll
    for (int i = 0; i < 8; ++i) { r[i] = lo[i]; r[8 + i] = hi[i]; }
    return r;
}
__device__ inline v16h load_frag_b(const _Float16* p) {
    v8h lo = *(const v8h*)(p);
    v8h hi = *(const v8h*)(p + 16);
    v16h r;
#pragma unroll
    for (int i = 0; i < 8; ++i) { r[i] = lo[i]; r[8 + i] = hi[i]; }
    return r;
}

// ---------------------------------------------------------------------------
// GEMM: trans[v][d] = sum_k diags[d][k] * emb[k][v] + bias[d], stored f16.
// Register-blocked: each wave computes a 32(M) x 64(N) region = 2x4 tiles of
// 16x16, so per K-step 2 A-frags + 4 B-frags feed 8 WMMAs (1.5 loads/WMMA).
// D 16x16 f32 layout: lane%16 = N, vgpr j -> M = j + 8*(lane/16), i.e. each
// lane holds 8 consecutive M (=d) values -> one contiguous 16B f16 store.
// ---------------------------------------------------------------------------
__global__ void __launch_bounds__(256) gemm_trans(const _Float16* __restrict__ Ah,
                                                  const _Float16* __restrict__ BhT,
                                                  const float* __restrict__ bias,
                                                  _Float16* __restrict__ transh) {
    const int wave  = threadIdx.x >> 5;
    const int lane  = threadIdx.x & 31;
    const int strip = blockIdx.x * 8 + wave;       // 64-wide N strip
    if (strip >= NSTRIPS) return;                  // wave-uniform, EXEC stays full
    const int n0 = strip * 64;
    const int m0 = blockIdx.y * 32;

    const int half = lane >> 4;
    const int sub  = lane & 15;

    const _Float16* ap0 = Ah + (size_t)(m0 + sub) * KPAD + half * 8;
    const _Float16* ap1 = ap0 + (size_t)16 * KPAD;
    const _Float16* bp[4];
#pragma unroll
    for (int nt = 0; nt < 4; ++nt)
        bp[nt] = BhT + (size_t)(n0 + nt * 16 + sub) * KPAD + half * 8;

    v8f acc[2][4] = {};
    for (int k0 = 0; k0 < KPAD; k0 += 32) {
        __builtin_prefetch(ap0 + k0 + 32, 0, 1);   // global_prefetch_b8
        __builtin_prefetch(ap1 + k0 + 32, 0, 1);
        v16h a0 = load_frag_a(ap0 + k0);
        v16h a1 = load_frag_a(ap1 + k0);
#pragma unroll
        for (int nt = 0; nt < 4; ++nt) {
            __builtin_prefetch(bp[nt] + k0 + 32, 0, 1);
            v16h b = load_frag_b(bp[nt] + k0);
            acc[0][nt] = __builtin_amdgcn_wmma_f32_16x16x32_f16(
                false, a0, false, b, (short)0, acc[0][nt], false, false);
            acc[1][nt] = __builtin_amdgcn_wmma_f32_16x16x32_f16(
                false, a1, false, b, (short)0, acc[1][nt], false, false);
        }
    }

    // Epilogue: + bias[d], convert to f16, store trans[v][d] (v-major).
#pragma unroll
    for (int mt = 0; mt < 2; ++mt) {
        const int mbase = m0 + mt * 16 + 8 * half; // 8 consecutive d values
        float bv[8];
#pragma unroll
        for (int j = 0; j < 8; ++j)
            bv[j] = (mbase + j < DDIM) ? bias[mbase + j] : 0.0f;
        const bool full = (mbase + 8 <= DDIM);
#pragma unroll
        for (int nt = 0; nt < 4; ++nt) {
            const int bcol = n0 + nt * 16 + sub;   // v (may land in pad rows)
            _Float16* orow = transh + (size_t)bcol * DDIM + mbase;
            if (full) {
                v8h o;
#pragma unroll
                for (int j = 0; j < 8; ++j)
                    o[j] = (_Float16)(acc[mt][nt][j] + bv[j]);
                *(v8h*)orow = o;                   // 16B contiguous store
            } else {
#pragma unroll
                for (int j = 0; j < 8; ++j)
                    if (mbase + j < DDIM)
                        orow[j] = (_Float16)(acc[mt][nt][j] + bv[j]);
            }
        }
    }
}

// ---------------------------------------------------------------------------
// Max-plus DP scan over T. One thread per (b, p); hiddens in registers.
// trans rows (84 MB f16 total) are L2-resident on MI455X (192 MB L2).
// ---------------------------------------------------------------------------
__global__ void __launch_bounds__(256) dp_scan(const int* __restrict__ docs,
                                               const int* __restrict__ doc_lens,
                                               const _Float16* __restrict__ transh,
                                               const float* __restrict__ epsilon,
                                               float* __restrict__ scores) {
    int tid = blockIdx.x * blockDim.x + threadIdx.x;
    if (tid >= BDIM * PDIM) return;
    int b = tid / PDIM, p = tid % PDIM;

    float eps[LDIM - 1];
#pragma unroll
    for (int l = 0; l < LDIM - 1; ++l) eps[l] = epsilon[p * (LDIM - 1) + l];

    float h[LDIM];
    h[0] = 0.0f;
#pragma unroll
    for (int l = 1; l < LDIM; ++l) h[l] = NEGV;
    float score = NEGV;

    const int dl = doc_lens[b];
    const int* drow = docs + b * TDIM;

    for (int t = 0; t < TDIM; ++t) {
        int v = drow[t];
        const _Float16* row = transh + (size_t)v * DDIM + p * (2 * LDIM);
        float tm0[LDIM], tm1[LDIM];
#pragma unroll
        for (int l = 0; l < LDIM; ++l) {
            tm0[l] = (float)row[l];          // diag 0
            tm1[l] = (float)row[LDIM + l];   // diag 1
        }
        float ae[LDIM];
        ae[0] = h[0];
#pragma unroll
        for (int l = 1; l < LDIM; ++l) ae[l] = fmaxf(h[l], h[l - 1] + eps[l - 1]);
        float nh[LDIM];
        nh[0] = fmaxf(0.0f, ae[0] + tm0[0]);        // one_pad vs self-loop
#pragma unroll
        for (int l = 1; l < LDIM; ++l)
            nh[l] = fmaxf(ae[l - 1] + tm1[l - 1], ae[l] + tm0[l]);
#pragma unroll
        for (int l = 0; l < LDIM; ++l) h[l] = nh[l];
        if (t < dl) score = fmaxf(score, h[LDIM - 1]);
    }
    scores[tid] = score;
}

// ---------------------------------------------------------------------------
// MLP head: out[b] = relu(scores[b] @ w0 + b0) @ w1 + b1. One block per b.
// ---------------------------------------------------------------------------
__global__ void __launch_bounds__(HDIM) mlp_head(const float* __restrict__ scores,
                                                 const float* __restrict__ w0,
                                                 const float* __restrict__ b0,
                                                 const float* __restrict__ w1,
                                                 const float* __restrict__ b1,
                                                 float* __restrict__ out) {
    __shared__ float hbuf[HDIM];
    const int b = blockIdx.x, j = threadIdx.x;
    const float* s = scores + b * PDIM;
    float acc = b0[j];
    for (int p = 0; p < PDIM; ++p) acc += s[p] * w0[p * HDIM + j];
    hbuf[j] = fmaxf(acc, 0.0f);
    __syncthreads();
    if (j < CDIM) {
        float o = b1[j];
        for (int k = 0; k < HDIM; ++k) o += hbuf[k] * w1[k * CDIM + j];
        out[b * CDIM + j] = o;
    }
}

// ---------------------------------------------------------------------------
extern "C" void kernel_launch(void* const* d_in, const int* in_sizes, int n_in,
                              void* d_out, int out_size, void* d_ws, size_t ws_size,
                              hipStream_t stream) {
    const int*   docs     = (const int*)d_in[0];
    const int*   doc_lens = (const int*)d_in[1];
    const float* emb      = (const float*)d_in[2];
    const float* diags    = (const float*)d_in[3];
    const float* bias     = (const float*)d_in[4];
    const float* epsilon  = (const float*)d_in[5];
    const float* w0       = (const float*)d_in[6];
    const float* b0       = (const float*)d_in[7];
    const float* w1       = (const float*)d_in[8];
    const float* b1       = (const float*)d_in[9];
    float* out = (float*)d_out;

    // Workspace layout (all offsets 256B-aligned):
    //   Ah     : MPAD*KPAD f16  =  2,703,360 B
    //   BhT    : NPAD*KPAD f16  =  6,430,720 B
    //   transh : NPAD*DDIM f16  = 84,403,200 B  (fits in MI455X 192MB L2)
    //   scores : B*P f32        =     76,800 B
    char* ws = (char*)d_ws;
    const size_t ahBytes = (size_t)MPAD * KPAD * 2;
    const size_t bhBytes = (size_t)NPAD * KPAD * 2;
    const size_t trBytes = (size_t)NPAD * DDIM * 2;
    _Float16* Ah     = (_Float16*)ws;
    _Float16* BhT    = (_Float16*)(ws + ahBytes);
    _Float16* transh = (_Float16*)(ws + ahBytes + bhBytes);
    float*    scores = (float*)(ws + ahBytes + bhBytes + trBytes);

    pack_a<<<(MPAD * KPAD + 255) / 256, 256, 0, stream>>>(diags, Ah);
    pack_b<<<(NPAD * KPAD + 255) / 256, 256, 0, stream>>>(emb, BhT);

    dim3 gg((NSTRIPS + 7) / 8, MSTRIPS);
    gemm_trans<<<gg, 256, 0, stream>>>(Ah, BhT, bias, transh);

    dp_scan<<<(BDIM * PDIM + 255) / 256, 256, 0, stream>>>(docs, doc_lens,
                                                           transh, epsilon, scores);

    mlp_head<<<BDIM, HDIM, 0, stream>>>(scores, w0, b0, w1, b1, out);
}
